// MultiheadAttention_56367150792793
// MI455X (gfx1250) — compile-verified, matrix operand order checked by reading, not run on hardware
//
#include <hip/hip_runtime.h>
#include <hip/hip_bf16.h>

typedef __attribute__((ext_vector_type(16))) _Float16 v16h;
typedef __attribute__((ext_vector_type(8)))  float    v8f;

// Problem constants: B=4, S=1024, D=1024, H=16, DEPTH=64
#define S_LEN  1024
#define D_DIM  1024
#define N_HEAD 16
#define DEPTH  64
#define M_ROWS 4096   // B*S

// ---- CDNA5 async global->LDS path (guarded; falls back to plain copies) ----
#if __has_builtin(__builtin_amdgcn_global_load_async_to_lds_b128) && \
    __has_builtin(__builtin_amdgcn_s_wait_asynccnt)
#define HAVE_ASYNC_LDS 1
typedef int v4i __attribute__((vector_size(16)));
typedef __attribute__((address_space(1))) v4i* gv4p;   // global int4*
typedef __attribute__((address_space(3))) v4i* lv4p;   // LDS int4*
#else
#define HAVE_ASYNC_LDS 0
#endif

static __device__ __forceinline__ v8f wmma32(v16h a, v16h b, v8f c) {
  // D = A(16x32 f16) * B(32x16 f16) + C(16x16 f32)
  return __builtin_amdgcn_wmma_f32_16x16x32_f16(
      /*neg_a=*/false, a, /*neg_b=*/false, b,
      /*c_mod=*/(short)0, c, /*reuse_a=*/false, /*reuse_b=*/false);
}

// Cross-half exchange: SWAPX16 (xor=0x10, and=0x1f) -> offset 0x401f. wave32.
static __device__ __forceinline__ float swz16(float x) {
  return __int_as_float(__builtin_amdgcn_ds_swizzle(__float_as_int(x), 0x401f));
}

// 16x32 fragment (A layout; identical indexing serves the B operand when the
// source is the transposed matrix).  rowptr -> this lane's row, K contiguous.
static __device__ __forceinline__ v16h frag_rowmajor(const _Float16* rowptr,
                                                     int koff) {
  const int half = (threadIdx.x >> 4) & 1;
  v16h r;
#pragma unroll
  for (int v = 0; v < 8; ++v) {
    int k0 = koff + (v >> 2) * 16 + half * 8 + (v & 3) * 2;
    r[2 * v]     = rowptr[k0];
    r[2 * v + 1] = rowptr[k0 + 1];
  }
  return r;
}

// ---------------------------------------------------------------------------
// Projection GEMM: Y = X @ W + bias.  Block tile 128x128, 8 waves each owning
// a 32x64 strip (2 A-frags x 4 B-frags = 8 WMMAs per 32-k chunk, B reused).
// OUT_MODE 0: f16 head-split [B,H,S,DEPTH]       (Q, K; scaled)
// OUT_MODE 1: f32 row-major  [M,N]               (final projection)
// OUT_MODE 2: f16 head-split transposed [B,H,DEPTH,S]  (V -> async-friendly)
// ---------------------------------------------------------------------------
template <typename TIn, int OUT_MODE>
__global__ __launch_bounds__(256) void proj_kernel(
    const TIn* __restrict__ X, const float* __restrict__ W,
    const float* __restrict__ bias, void* __restrict__ Yv, float scale) {
  __shared__ __align__(16) _Float16 Xl[128 * 34];  // A tile, padded stride
  __shared__ __align__(16) _Float16 WT[128 * 34];  // W tile transposed [n][k]

  const int tid  = threadIdx.x;
  const int lane = tid & 31;
  const int wave = tid >> 5;
  const int wm   = wave & 3;   // 4 M-groups of 32
  const int wn   = wave >> 2;  // 2 N-groups of 64
  const int mBase = blockIdx.y * 128;
  const int nBase = blockIdx.x * 128;

  v8f acc[2][4] = {};

  for (int kc = 0; kc < D_DIM; kc += 32) {
    __syncthreads();
#pragma unroll
    for (int e = 0; e < 16; ++e) {         // stage X: 128x32
      int id = tid + 256 * e;
      int r = id >> 5, k = id & 31;
      Xl[r * 34 + k] =
          (_Float16)(float)X[(size_t)(mBase + r) * D_DIM + kc + k];
    }
#pragma unroll
    for (int e = 0; e < 16; ++e) {         // stage W^T: 32x128 -> [n][k]
      int id = tid + 256 * e;
      int kk = id >> 7, n = id & 127;
      WT[n * 34 + kk] = (_Float16)W[(size_t)(kc + kk) * D_DIM + nBase + n];
    }
    __syncthreads();

    const int r = lane & 15;
    v16h a0 = frag_rowmajor(&Xl[(wm * 32 + r) * 34], 0);
    v16h a1 = frag_rowmajor(&Xl[(wm * 32 + 16 + r) * 34], 0);
#pragma unroll
    for (int j = 0; j < 4; ++j) {
      v16h b = frag_rowmajor(&WT[(wn * 64 + j * 16 + r) * 34], 0);
      acc[0][j] = wmma32(a0, b, acc[0][j]);
      acc[1][j] = wmma32(a1, b, acc[1][j]);
    }
  }

  // Epilogue: C layout -> lane (0-15)=N, regs=M rows (high half +8)
  const int half = lane >> 4;
  const int col  = lane & 15;
#pragma unroll
  for (int t = 0; t < 2; ++t)
#pragma unroll
    for (int j = 0; j < 4; ++j) {
      const int n = nBase + wn * 64 + j * 16 + col;
      const float bv = bias[n];
#pragma unroll
      for (int i = 0; i < 8; ++i) {
        const int mg = mBase + wm * 32 + t * 16 + i + half * 8;
        const float val = (acc[t][j][i] + bv) * scale;
        const int bb = mg >> 10, s = mg & (S_LEN - 1);
        const int hh = n >> 6,  d = n & (DEPTH - 1);
        if (OUT_MODE == 0) {
          ((_Float16*)Yv)[(((size_t)bb * N_HEAD + hh) * S_LEN + s) * DEPTH + d] =
              (_Float16)val;
        } else if (OUT_MODE == 2) {
          ((_Float16*)Yv)[(((size_t)bb * N_HEAD + hh) * DEPTH + d) * S_LEN + s] =
              (_Float16)val;
        } else {
          ((float*)Yv)[(size_t)mg * D_DIM + n] = val;
        }
      }
    }
}

// ---------------------------------------------------------------------------
// Flash attention.  Q,K head-split [B,H,S,64] f16; V head-split TRANSPOSED
// [B,H,64,S] f16.  Per block: 8 waves x 16 q rows = 128 queries of one (b,h).
// Computes S^T = K*Q^T so the exp'd score fragment is already in B-operand
// layout for ctx^T = V^T * P^T (no cross-lane repack).  Q pre-scaled by 1/8.
// K/V tiles staged to LDS with GLOBAL_LOAD_ASYNC_TO_LDS_B128 when available.
// ---------------------------------------------------------------------------
__global__ __launch_bounds__(256) void flash_kernel(
    const _Float16* __restrict__ Q, const _Float16* __restrict__ K,
    const _Float16* __restrict__ V, const float* __restrict__ mask,
    _Float16* __restrict__ ctx) {
  __shared__ __align__(16) _Float16 Kt[32 * 72];  // [k][d], stride 72 halfs
  __shared__ __align__(16) _Float16 Vt[64 * 40];  // [d][k], stride 40 halfs
  __shared__ float mS[32];

  const int tid  = threadIdx.x;
  const int lane = tid & 31;
  const int wave = tid >> 5;
  const int half = lane >> 4;
  const int col  = lane & 15;

  const int bh = blockIdx.y;               // b*16 + h
  const int b  = bh >> 4, h = bh & 15;
  const size_t headOff = (size_t)bh * S_LEN * DEPTH;
  const _Float16* Qp = Q + headOff;
  const _Float16* Kp = K + headOff;
  const _Float16* Vp = V + headOff;        // transposed: row d, stride S_LEN
  const float* maskp = mask + (size_t)b * S_LEN;

  const int qBase = blockIdx.x * 128 + wave * 16;

  // Q^T as B operand: lane=q column, contiguous depth pairs from global.
  const _Float16* qrow = Qp + (size_t)(qBase + col) * DEPTH;
  v16h bQ0 = frag_rowmajor(qrow, 0);
  v16h bQ1 = frag_rowmajor(qrow, 32);

  float mPrev = -1e30f, l = 0.0f;
  v8f o[4] = {};

  for (int t = 0; t < S_LEN / 32; ++t) {
    const int kB = t * 32;
    __syncthreads();
#if HAVE_ASYNC_LDS
    {
      // K tile: 32 rows x 64 halfs = 256 x 16B
      const int kk = tid >> 3, seg = (tid & 7) * 8;
      __builtin_amdgcn_global_load_async_to_lds_b128(
          (gv4p)(Kp + (size_t)(kB + kk) * DEPTH + seg),
          (lv4p)&Kt[kk * 72 + seg], 0, 0);
      // V^T tile: 64 rows x 32 halfs = 256 x 16B
      const int dd = tid >> 2, sg = (tid & 3) * 8;
      __builtin_amdgcn_global_load_async_to_lds_b128(
          (gv4p)(Vp + (size_t)dd * S_LEN + kB + sg),
          (lv4p)&Vt[dd * 40 + sg], 0, 0);
      if (tid < 32) mS[tid] = maskp[kB + tid];
      __builtin_amdgcn_s_wait_asynccnt(0);
    }
#else
#pragma unroll
    for (int e = 0; e < 8; ++e) {
      int id = tid + 256 * e;
      int kk = id >> 6, d = id & 63;       // K: [k][d]
      Kt[kk * 72 + d] = Kp[(size_t)(kB + kk) * DEPTH + d];
      int dv = id >> 5, k2 = id & 31;      // V^T: [d][k]
      Vt[dv * 40 + k2] = Vp[(size_t)dv * S_LEN + kB + k2];
    }
    if (tid < 32) mS[tid] = maskp[kB + tid];
#endif
    __syncthreads();

    if (t + 1 < S_LEN / 32 && lane == 0) {  // global_prefetch_b8 next tiles
      __builtin_prefetch(Kp + (size_t)(kB + 32) * DEPTH, 0, 1);
      __builtin_prefetch(Vp + kB + 32, 0, 1);
    }

    const int r = col;
    // S^T fragments: c0 = keys kB..kB+15, c1 = keys kB+16..kB+31
    v8f c0 = {}, c1 = {};
    c0 = wmma32(frag_rowmajor(&Kt[r * 72], 0),         bQ0, c0);
    c0 = wmma32(frag_rowmajor(&Kt[r * 72], 32),        bQ1, c0);
    c1 = wmma32(frag_rowmajor(&Kt[(16 + r) * 72], 0),  bQ0, c1);
    c1 = wmma32(frag_rowmajor(&Kt[(16 + r) * 72], 32), bQ1, c1);

    // mask (depends only on key index) + online softmax over k
    float mx = -1e30f;
#pragma unroll
    for (int i = 0; i < 8; ++i) {
      c0[i] += mS[i + half * 8] * (-1e9f);
      c1[i] += mS[16 + i + half * 8] * (-1e9f);
      mx = fmaxf(mx, fmaxf(c0[i], c1[i]));
    }
    mx = fmaxf(mx, swz16(mx));             // combine k-halves across lanes
    const float mNew = fmaxf(mPrev, mx);
    const float sc = __expf(mPrev - mNew);

    float p0[8], p1[8], rs = 0.0f;
#pragma unroll
    for (int i = 0; i < 8; ++i) {
      p0[i] = __expf(c0[i] - mNew);
      p1[i] = __expf(c1[i] - mNew);
      rs += p0[i] + p1[i];
    }
    rs += swz16(rs);
    l = l * sc + rs;
    mPrev = mNew;

    // P^T fragment lands directly in B layout: v0-3 from c0, v4-7 from c1.
    v16h bP;
#pragma unroll
    for (int i = 0; i < 8; ++i) {
      bP[i]     = (_Float16)p0[i];
      bP[8 + i] = (_Float16)p1[i];
    }

#pragma unroll
    for (int f = 0; f < 4; ++f) {
#pragma unroll
      for (int i = 0; i < 8; ++i) o[f][i] *= sc;   // rescale accumulators
      o[f] = wmma32(frag_rowmajor(&Vt[(f * 16 + r) * 40], 0), bP, o[f]);
    }
  }

  // Epilogue: ctx^T frag -> ctx[b][s][h*64+d] (f16 workspace)
  const float inv = 1.0f / l;
  const int s = qBase + col;
  _Float16* crow = ctx + ((size_t)b * S_LEN + s) * D_DIM + h * DEPTH;
#pragma unroll
  for (int f = 0; f < 4; ++f)
#pragma unroll
    for (int i = 0; i < 8; ++i) {
      const int d = f * 16 + i + half * 8;
      crow[d] = (_Float16)(o[f][i] * inv);
    }
}

// ---------------------------------------------------------------------------
extern "C" void kernel_launch(void* const* d_in, const int* in_sizes, int n_in,
                              void* d_out, int out_size, void* d_ws,
                              size_t ws_size, hipStream_t stream) {
  (void)in_sizes; (void)n_in; (void)out_size; (void)ws_size;
  const float* query  = (const float*)d_in[0];
  const float* key_in = (const float*)d_in[1];
  const float* value  = (const float*)d_in[2];
  const float* mask   = (const float*)d_in[3];
  const float* wq = (const float*)d_in[4];  const float* bq = (const float*)d_in[5];
  const float* wk = (const float*)d_in[6];  const float* bk = (const float*)d_in[7];
  const float* wv = (const float*)d_in[8];  const float* bv = (const float*)d_in[9];
  const float* wo = (const float*)d_in[10]; const float* bo = (const float*)d_in[11];

  const size_t headElems = (size_t)4 * N_HEAD * S_LEN * DEPTH;  // 4 Mi elems
  _Float16* qws = (_Float16*)d_ws;
  _Float16* kws = qws + headElems;
  _Float16* vws = kws + headElems;     // V stored transposed [B,H,DEPTH,S]
  _Float16* cws = vws + headElems;     // ctx f16, [B,S,D]; total 32 MB

  dim3 blk(256);
  dim3 gProj(D_DIM / 128, M_ROWS / 128);    // 8 x 32
  dim3 gFlash(S_LEN / 128, 4 * N_HEAD);     // 8 x 64

  // Q gets 1/sqrt(DEPTH) folded into its projection output.
  proj_kernel<float, 0><<<gProj, blk, 0, stream>>>(query,  wq, bq, qws, 0.125f);
  proj_kernel<float, 0><<<gProj, blk, 0, stream>>>(key_in, wk, bk, kws, 1.0f);
  proj_kernel<float, 2><<<gProj, blk, 0, stream>>>(value,  wv, bv, vws, 1.0f);
  flash_kernel<<<gFlash, blk, 0, stream>>>(qws, kws, vws, mask, cws);
  proj_kernel<_Float16, 1><<<gProj, blk, 0, stream>>>(cws, wo, bo, d_out, 1.0f);
}
